// TemplateEmbeddingIteration_61400852464444
// MI455X (gfx1250) — matrix operand order repaired and checked by reading, not compiled, not run on hardware
//
#include <hip/hip_runtime.h>

typedef __attribute__((ext_vector_type(16))) __bf16 v16bf;
typedef __attribute__((ext_vector_type(8)))  float  v8f;

#define NN   256
#define CC   64
#define RTOT 65536   // NN*NN
#define HH   4
#define CA   16

static __device__ inline v8f vzero() {
  v8f z = {0.f,0.f,0.f,0.f,0.f,0.f,0.f,0.f};
  return z;
}

static __device__ inline v8f wmma_bf16(v16bf a, v16bf b, v8f c) {
  return __builtin_amdgcn_wmma_f32_16x16x32_bf16(false, a, false, b, (short)0, c, false, false);
}

// ---- Fast transcendentals (hardware v_exp_f32 / v_rcp_f32) ---------------
static __device__ inline float fast_exp(float x) { return __expf(x); }
static __device__ inline float fast_rcp(float x) { return __builtin_amdgcn_rcpf(x); }
static __device__ inline float sigmoidf(float x) { return fast_rcp(1.f + fast_exp(-x)); }

// ---- Fragment loaders (contiguous K order K = 16*(lane/16)+e, wave32) -------------
// A: 16x32 bf16, row-major source, stride ld. lane<16: row=lane, K=e;
// lane>=16: row=lane-16, K=16+e.
static __device__ inline v16bf load_a(const __bf16* __restrict__ src, int ld) {
  int lane = threadIdx.x & 31;
  int row  = lane & 15;
  int kofs = (lane >> 4) << 4;
  const __bf16* p = src + row * ld;
  v16bf a;
#pragma unroll
  for (int e = 0; e < 16; ++e) {
    int k = e + kofs;
    a[e] = p[k];
  }
  return a;
}

// A with only 16 real K values (K 16..31 are zero)
static __device__ inline v16bf load_a_pad16(const __bf16* __restrict__ src, int ld) {
  int lane = threadIdx.x & 31;
  int row  = lane & 15;
  int kofs = (lane >> 4) << 4;
  const __bf16* p = src + row * ld;
  v16bf a;
#pragma unroll
  for (int e = 0; e < 16; ++e) {
    int k = e + kofs;
    a[e] = (k < 16) ? p[k] : (__bf16)0.f;
  }
  return a;
}

// B: 32x16 bf16 from K-major source B[k][n], stride ld.
// lane<16: col=lane, K=e; lane>=16: col=lane-16, K=16+e.
static __device__ inline v16bf load_b_kn(const __bf16* __restrict__ src, int ld) {
  int lane = threadIdx.x & 31;
  int col  = lane & 15;
  int kofs = (lane >> 4) << 4;
  v16bf b;
#pragma unroll
  for (int e = 0; e < 16; ++e) b[e] = src[(kofs + e) * ld + col];
  return b;
}

// B from N-major source src[n][k] (stride ld): B[k][n] = src[n][k]
static __device__ inline v16bf load_b_nk(const __bf16* __restrict__ src, int ld) {
  int lane = threadIdx.x & 31;
  int col  = lane & 15;
  int kofs = (lane >> 4) << 4;
  const __bf16* p = src + col * ld + kofs;
  v16bf b;
#pragma unroll
  for (int e = 0; e < 16; ++e) b[e] = p[e];
  return b;
}

// B from N-major bf16 source, only 16 real K (lanes>=16 zero)
static __device__ inline v16bf load_b_nk_pad16(const __bf16* __restrict__ src, int ld) {
  int lane = threadIdx.x & 31;
  int col  = lane & 15;
  int hi   = lane >> 4;
  const __bf16* p = src + col * ld;
  v16bf b;
#pragma unroll
  for (int e = 0; e < 16; ++e) b[e] = hi ? (__bf16)0.f : p[e];
  return b;
}

// ---- Weight f32 -> bf16 arena conversion ---------------------------------
__global__ void cvt_kernel(const float* __restrict__ s, __bf16* __restrict__ d) {
  int i = blockIdx.x * 256 + threadIdx.x;
  d[i] = (__bf16)s[i];
}

// ---- LayerNorm over last dim (64), f32 -> bf16; optional transposed read -
__global__ void ln_kernel(const float* __restrict__ x, const float* __restrict__ g,
                          const float* __restrict__ b, __bf16* __restrict__ out, int t) {
  int wave = threadIdx.x >> 5, lane = threadIdx.x & 31;
  int row = blockIdx.x * 8 + wave;
  int src = t ? ((row & 255) * NN + (row >> 8)) : row;
  float2 v = *(const float2*)(x + (size_t)src * CC + lane * 2);
  float s = v.x + v.y;
#pragma unroll
  for (int m = 1; m < 32; m <<= 1) s += __shfl_xor(s, m, 32);
  float mean = s * (1.f / 64.f);
  float dx = v.x - mean, dy = v.y - mean;
  float vs = dx * dx + dy * dy;
#pragma unroll
  for (int m = 1; m < 32; m <<= 1) vs += __shfl_xor(vs, m, 32);
  float inv = rsqrtf(vs * (1.f / 64.f) + 1e-5f);
  out[(size_t)row * CC + lane * 2]     = (__bf16)(dx * inv * g[2 * lane]     + b[2 * lane]);
  out[(size_t)row * CC + lane * 2 + 1] = (__bf16)(dy * inv * g[2 * lane + 1] + b[2 * lane + 1]);
}

// ---- Triangle-mult projections: left/right (gated+masked) and gate -------
// Weights are bf16 arena: lp,lg,rp,rg,g at fixed 4096-element slots.
__global__ void tm_proj_kernel(const __bf16* __restrict__ a, const float* __restrict__ mask,
                               const __bf16* __restrict__ W,   // arena base for this block
                               const float* lp_b, const float* lg_b,
                               const float* rp_b, const float* rg_b,
                               const float* g_b,
                               __bf16* __restrict__ left, __bf16* __restrict__ right,
                               float* __restrict__ gate, int outgoing) {
  const __bf16* lp_w = W;
  const __bf16* lg_w = W + 4096;
  const __bf16* rp_w = W + 8192;
  const __bf16* rg_w = W + 12288;
  const __bf16* g_w  = W + 16384;
  int wave = threadIdx.x >> 5;
  int rt = blockIdx.x * 8 + wave;
  int row0 = rt * 16;
  v16bf a0 = load_a(a + (size_t)row0 * CC,      CC);
  v16bf a1 = load_a(a + (size_t)row0 * CC + 32, CC);
  int lane = threadIdx.x & 31, col = lane & 15, rb = (lane >> 4) << 3;

#pragma unroll
  for (int side = 0; side < 2; ++side) {
    const __bf16* pw = side ? rp_w : lp_w; const float* pb = side ? rp_b : lp_b;
    const __bf16* gw = side ? rg_w : lg_w; const float* gb = side ? rg_b : lg_b;
    __bf16* dst = side ? right : left;
#pragma unroll
    for (int ct = 0; ct < 4; ++ct) {
      v8f accg = vzero(), accp = vzero();
      accg = wmma_bf16(a0, load_b_nk(gw + ct * 16 * CC,      CC), accg);
      accg = wmma_bf16(a1, load_b_nk(gw + ct * 16 * CC + 32, CC), accg);
      accp = wmma_bf16(a0, load_b_nk(pw + ct * 16 * CC,      CC), accp);
      accp = wmma_bf16(a1, load_b_nk(pw + ct * 16 * CC + 32, CC), accp);
      int c = ct * 16 + col;
      float bp = pb[c], bg = gb[c];
#pragma unroll
      for (int r = 0; r < 8; ++r) {
        int pos = row0 + rb + r;
        float val = mask[pos] * (accp[r] + bp) * sigmoidf(accg[r] + bg);
        int x = pos >> 8, y = pos & 255;
        int idx = (c << 16) + (outgoing ? (x << 8) + y : (y << 8) + x);
        dst[idx] = (__bf16)val;
      }
    }
  }
  // gate (pre-sigmoid), row-major [pos][c]
#pragma unroll
  for (int ct = 0; ct < 4; ++ct) {
    v8f acc = vzero();
    acc = wmma_bf16(a0, load_b_nk(g_w + ct * 16 * CC,      CC), acc);
    acc = wmma_bf16(a1, load_b_nk(g_w + ct * 16 * CC + 32, CC), acc);
    int c = ct * 16 + col;
    float bb = g_b[c];
#pragma unroll
    for (int r = 0; r < 8; ++r) {
      int pos = row0 + rb + r;
      gate[(size_t)pos * CC + c] = acc[r] + bb;
    }
  }
}

// ---- Triangle einsum: out[i,j,c] = sum_k A[c][i][k] * B[c][j][k] ---------
__global__ void tri_einsum_kernel(const __bf16* __restrict__ Ab, const __bf16* __restrict__ Bb,
                                  float* __restrict__ pp) {
  int wave = threadIdx.x >> 5;
  int gid = blockIdx.x * 8 + wave;                  // 16384 tiles
  int c = gid >> 8, it = (gid >> 4) & 15, jt = gid & 15;
  const __bf16* ab = Ab + (size_t)c * RTOT + it * 16 * NN;
  const __bf16* bb = Bb + (size_t)c * RTOT + jt * 16 * NN;
  v8f acc = vzero();
#pragma unroll
  for (int kt = 0; kt < 8; ++kt)
    acc = wmma_bf16(load_a(ab + kt * 32, NN), load_b_nk(bb + kt * 32, NN), acc);
  int lane = threadIdx.x & 31, col = lane & 15, rb = (lane >> 4) << 3;
#pragma unroll
  for (int r = 0; r < 8; ++r) {
    int i = it * 16 + rb + r, j = jt * 16 + col;
    pp[((size_t)i * NN + j) * CC + c] = acc[r];
  }
}

// ---- Gated output projection of triangle mult: act += (LNpp@W + b)*sig(gate)
__global__ void gated_out_kernel(const __bf16* __restrict__ a, const __bf16* __restrict__ W,
                                 const float* __restrict__ bias, const float* __restrict__ gate,
                                 float* __restrict__ act) {
  int wave = threadIdx.x >> 5;
  int row0 = (blockIdx.x * 8 + wave) * 16;
  v16bf a0 = load_a(a + (size_t)row0 * CC,      CC);
  v16bf a1 = load_a(a + (size_t)row0 * CC + 32, CC);
  int lane = threadIdx.x & 31, col = lane & 15, rb = (lane >> 4) << 3;
#pragma unroll
  for (int ct = 0; ct < 4; ++ct) {
    v8f acc = vzero();
    acc = wmma_bf16(a0, load_b_nk(W + ct * 16 * CC,      CC), acc);
    acc = wmma_bf16(a1, load_b_nk(W + ct * 16 * CC + 32, CC), acc);
    int c = ct * 16 + col;
    float bb = bias[c];
#pragma unroll
    for (int r = 0; r < 8; ++r) {
      size_t idx = ((size_t)(row0 + rb + r)) * CC + c;
      act[idx] += (acc[r] + bb) * sigmoidf(gate[idx]);
    }
  }
}

// ---- nb[h][i*N+j] = sum_c a[i,j,c] * f2d[h,c] ----------------------------
__global__ void nb_kernel(const __bf16* __restrict__ a, const float* __restrict__ f2d,
                          float* __restrict__ nb) {
  int wave = threadIdx.x >> 5, lane = threadIdx.x & 31;
  int row = blockIdx.x * 8 + wave;
  float a0 = (float)a[(size_t)row * CC + 2 * lane];
  float a1 = (float)a[(size_t)row * CC + 2 * lane + 1];
#pragma unroll
  for (int h = 0; h < HH; ++h) {
    float p = a0 * f2d[h * CC + 2 * lane] + a1 * f2d[h * CC + 2 * lane + 1];
#pragma unroll
    for (int m = 1; m < 32; m <<= 1) p += __shfl_xor(p, m, 32);
    if (lane == 0) nb[(size_t)h * RTOT + row] = p;
  }
}

// ---- Attention projections: q(scaled)/k/v bf16 + gate sigmoid f32 --------
// Arena slots: wq, wk, wv, wg at +0, +4096, +8192, +12288.
__global__ void attn_proj_kernel(const __bf16* __restrict__ a,
                                 const __bf16* __restrict__ W, const float* bg,
                                 __bf16* __restrict__ q, __bf16* __restrict__ k,
                                 __bf16* __restrict__ v, float* __restrict__ gs) {
  int wave = threadIdx.x >> 5;
  int row0 = (blockIdx.x * 8 + wave) * 16;
  v16bf a0 = load_a(a + (size_t)row0 * CC,      CC);
  v16bf a1 = load_a(a + (size_t)row0 * CC + 32, CC);
  int lane = threadIdx.x & 31, col = lane & 15, rb = (lane >> 4) << 3;
#pragma unroll
  for (int which = 0; which < 3; ++which) {
    const __bf16* Wx = W + which * 4096;
    __bf16* dst = which == 0 ? q : (which == 1 ? k : v);
    float scale = which == 0 ? 0.25f : 1.f;   // CA^-0.5
#pragma unroll
    for (int ct = 0; ct < 4; ++ct) {
      v8f acc = vzero();
      acc = wmma_bf16(a0, load_b_nk(Wx + ct * 16 * CC,      CC), acc);
      acc = wmma_bf16(a1, load_b_nk(Wx + ct * 16 * CC + 32, CC), acc);
      int c = ct * 16 + col;
#pragma unroll
      for (int r = 0; r < 8; ++r)
        dst[((size_t)(row0 + rb + r)) * CC + c] = (__bf16)(acc[r] * scale);
    }
  }
  const __bf16* Wg = W + 3 * 4096;
#pragma unroll
  for (int ct = 0; ct < 4; ++ct) {
    v8f acc = vzero();
    acc = wmma_bf16(a0, load_b_nk(Wg + ct * 16 * CC,      CC), acc);
    acc = wmma_bf16(a1, load_b_nk(Wg + ct * 16 * CC + 32, CC), acc);
    int c = ct * 16 + col;
    float bb = bg[c];
#pragma unroll
    for (int r = 0; r < 8; ++r)
      gs[((size_t)(row0 + rb + r)) * CC + c] = sigmoidf(acc[r] + bb);
  }
}

// ---- Attention core: one block per (i,h); 4 waves x 4 q-tiles ------------
__global__ void attn_kernel(const __bf16* __restrict__ q, const __bf16* __restrict__ k,
                            const __bf16* __restrict__ v, const float* __restrict__ gs,
                            const float* __restrict__ nb, const float* __restrict__ mask,
                            __bf16* __restrict__ obuf, int t) {
  __shared__ __bf16 lds_p[4][16 * NN];     // 32 KB: per-wave prob tiles
  int i = blockIdx.x >> 2, h = blockIdx.x & 3;
  int wave = threadIdx.x >> 5, lane = threadIdx.x & 31;
  int col = lane & 15, rb = (lane >> 4) << 3;
  const __bf16* qh = q + ((size_t)i * NN) * CC + h * CA;
  const __bf16* kh = k + ((size_t)i * NN) * CC + h * CA;
  const __bf16* vh = v + ((size_t)i * NN) * CC + h * CA;

  for (int qi = 0; qi < 4; ++qi) {
    int qt = wave * 4 + qi;
    v16bf qf = load_a_pad16(qh + (size_t)qt * 16 * CC, CC);
    v8f lfr[16];
#pragma unroll
    for (int jt = 0; jt < 16; ++jt) {
      v8f acc = vzero();
      acc = wmma_bf16(qf, load_b_nk_pad16(kh + (size_t)jt * 16 * CC, CC), acc);
      int j = jt * 16 + col;
      float mv = t ? mask[(size_t)j * NN + i] : mask[(size_t)i * NN + j];
      float bias = 1e9f * (mv - 1.f) + nb[(size_t)h * RTOT + (size_t)i * NN + j];
#pragma unroll
      for (int e = 0; e < 8; ++e) acc[e] += bias;
      lfr[jt] = acc;
    }
    // softmax over 256 j (16 cols/lane-group x 16 frags)
    float mx[8], sm[8];
#pragma unroll
    for (int e = 0; e < 8; ++e) mx[e] = -1e30f;
#pragma unroll
    for (int jt = 0; jt < 16; ++jt)
#pragma unroll
      for (int e = 0; e < 8; ++e) mx[e] = fmaxf(mx[e], lfr[jt][e]);
#pragma unroll
    for (int m = 1; m < 16; m <<= 1)
#pragma unroll
      for (int e = 0; e < 8; ++e) mx[e] = fmaxf(mx[e], __shfl_xor(mx[e], m, 32));
#pragma unroll
    for (int e = 0; e < 8; ++e) sm[e] = 0.f;
#pragma unroll
    for (int jt = 0; jt < 16; ++jt)
#pragma unroll
      for (int e = 0; e < 8; ++e) {
        float p = fast_exp(lfr[jt][e] - mx[e]);
        lfr[jt][e] = p;
        sm[e] += p;
      }
#pragma unroll
    for (int m = 1; m < 16; m <<= 1)
#pragma unroll
      for (int e = 0; e < 8; ++e) sm[e] += __shfl_xor(sm[e], m, 32);
#pragma unroll
    for (int e = 0; e < 8; ++e) sm[e] = fast_rcp(sm[e]);
#pragma unroll
    for (int jt = 0; jt < 16; ++jt)
#pragma unroll
      for (int e = 0; e < 8; ++e)
        lds_p[wave][(e + rb) * NN + jt * 16 + col] = (__bf16)(lfr[jt][e] * sm[e]);
    // same-wave LDS write -> read hazard: wait for DS ops to land
    asm volatile("s_wait_dscnt 0" ::: "memory");
    // o = P @ V  (K = 256)
    v8f ov = vzero();
#pragma unroll
    for (int kt = 0; kt < 8; ++kt)
      ov = wmma_bf16(load_a(&lds_p[wave][kt * 32], NN),
                     load_b_kn(vh + (size_t)kt * 32 * CC, CC), ov);
#pragma unroll
    for (int e = 0; e < 8; ++e) {
      size_t gidx = ((size_t)i * NN + qt * 16 + e + rb) * CC + h * CA + col;
      obuf[gidx] = (__bf16)(ov[e] * gs[gidx]);
    }
  }
}

// ---- Attention output projection: act[dst] += o@Wo^T + bo (t: transposed)
__global__ void attn_out_kernel(const __bf16* __restrict__ o, const __bf16* __restrict__ W,
                                const float* __restrict__ bias, float* __restrict__ act, int t) {
  int wave = threadIdx.x >> 5;
  int row0 = (blockIdx.x * 8 + wave) * 16;
  v16bf a0 = load_a(o + (size_t)row0 * CC,      CC);
  v16bf a1 = load_a(o + (size_t)row0 * CC + 32, CC);
  int lane = threadIdx.x & 31, col = lane & 15, rb = (lane >> 4) << 3;
#pragma unroll
  for (int ct = 0; ct < 4; ++ct) {
    v8f acc = vzero();
    acc = wmma_bf16(a0, load_b_nk(W + ct * 16 * CC,      CC), acc);
    acc = wmma_bf16(a1, load_b_nk(W + ct * 16 * CC + 32, CC), acc);
    int c = ct * 16 + col;
    float bb = bias[c];
#pragma unroll
    for (int r = 0; r < 8; ++r) {
      int pos = row0 + rb + r;
      int x = pos >> 8, y = pos & 255;
      int dst = t ? (y << 8) + x : pos;
      act[(size_t)dst * CC + c] += acc[r] + bb;
    }
  }
}

// ---- Transition: hid = relu(LNact @ W1 + b1)  (64 -> 128) ----------------
__global__ void trans1_kernel(const __bf16* __restrict__ a, const __bf16* __restrict__ W,
                              const float* __restrict__ bias, __bf16* __restrict__ hid) {
  int wave = threadIdx.x >> 5;
  int row0 = (blockIdx.x * 8 + wave) * 16;
  v16bf a0 = load_a(a + (size_t)row0 * CC,      CC);
  v16bf a1 = load_a(a + (size_t)row0 * CC + 32, CC);
  int lane = threadIdx.x & 31, col = lane & 15, rb = (lane >> 4) << 3;
#pragma unroll
  for (int ct = 0; ct < 8; ++ct) {
    v8f acc = vzero();
    acc = wmma_bf16(a0, load_b_nk(W + ct * 16 * CC,      CC), acc);
    acc = wmma_bf16(a1, load_b_nk(W + ct * 16 * CC + 32, CC), acc);
    int c = ct * 16 + col;
    float bb = bias[c];
#pragma unroll
    for (int r = 0; r < 8; ++r)
      hid[((size_t)(row0 + rb + r)) * 128 + c] = (__bf16)fmaxf(acc[r] + bb, 0.f);
  }
}

// ---- Transition: act += hid @ W2 + b2  (128 -> 64) -----------------------
__global__ void trans2_kernel(const __bf16* __restrict__ hid, const __bf16* __restrict__ W,
                              const float* __restrict__ bias, float* __restrict__ act) {
  int wave = threadIdx.x >> 5;
  int row0 = (blockIdx.x * 8 + wave) * 16;
  v16bf af[4];
#pragma unroll
  for (int kt = 0; kt < 4; ++kt) af[kt] = load_a(hid + (size_t)row0 * 128 + kt * 32, 128);
  int lane = threadIdx.x & 31, col = lane & 15, rb = (lane >> 4) << 3;
#pragma unroll
  for (int ct = 0; ct < 4; ++ct) {
    v8f acc = vzero();
#pragma unroll
    for (int kt = 0; kt < 4; ++kt)
      acc = wmma_bf16(af[kt], load_b_nk(W + ct * 16 * 128 + kt * 32, 128), acc);
    int c = ct * 16 + col;
    float bb = bias[c];
#pragma unroll
    for (int r = 0; r < 8; ++r)
      act[((size_t)(row0 + rb + r)) * CC + c] += acc[r] + bb;
  }
}

// ---- Host orchestration --------------------------------------------------
extern "C" void kernel_launch(void* const* d_in, const int* in_sizes, int n_in,
                              void* d_out, int out_size, void* d_ws, size_t ws_size,
                              hipStream_t stream) {
  (void)in_sizes; (void)n_in; (void)out_size; (void)ws_size;
  const size_t SZ_BF = (size_t)RTOT * CC * 2;   // 8 MB
  const size_t SZ_F  = (size_t)RTOT * CC * 4;   // 16 MB
  char* ws = (char*)d_ws;
  __bf16* a_bf  = (__bf16*)(ws);                         // LN output
  __bf16* buf1  = (__bf16*)(ws + SZ_BF);                 // left / q
  __bf16* buf2  = (__bf16*)(ws + 2 * SZ_BF);             // right / k
  __bf16* buf3  = (__bf16*)(ws + 3 * SZ_BF);             // v
  float*  gatef = (float*) (ws + 4 * SZ_BF);             // gate / gsig
  float*  ppf   = (float*) (ws + 4 * SZ_BF + SZ_F);      // pp f32 / hid bf16
  __bf16* hid   = (__bf16*)(ws + 4 * SZ_BF + SZ_F);
  float*  nbf   = (float*) (ws + 4 * SZ_BF + 2 * SZ_F);  // nb: 1 MB
  __bf16* obuf  = (__bf16*)(ws + 4 * SZ_BF + 2 * SZ_F + (size_t)HH * RTOT * 4);
  __bf16* wa    = (__bf16*)(ws + 4 * SZ_BF + 2 * SZ_F + (size_t)HH * RTOT * 4 + SZ_BF);

  const float* mask = (const float*)d_in[1];
  float* act = (float*)d_out;
  (void)hipMemcpyAsync(act, d_in[0], SZ_F, hipMemcpyDeviceToDevice, stream);

  auto F = [&](int idx) { return (const float*)d_in[idx]; };

  // ---- convert all GEMM weights to a bf16 arena (once per call) ----------
  // Arena layout (element offsets):
  //   tm_out: lp,lg,rp,rg,g,op  @ 0      (6*4096)
  //   tm_in : same              @ 24576
  //   ta_s  : wq,wk,wv,wg,wo    @ 49152  (5*4096)
  //   ta_e  : same              @ 69632
  //   tr    : w1 (8192), w2 (8192) @ 90112
  auto CV = [&](int src_idx, size_t off, int n) {
    cvt_kernel<<<n / 256, 256, 0, stream>>>(F(src_idx), wa + off);
  };
  for (int pass = 0; pass < 2; ++pass) {
    int b = pass == 0 ? 2 : 18;
    size_t o = pass * 24576;
    CV(b + 2,  o + 0,     4096);  // lp_w
    CV(b + 6,  o + 4096,  4096);  // lg_w
    CV(b + 4,  o + 8192,  4096);  // rp_w
    CV(b + 8,  o + 12288, 4096);  // rg_w
    CV(b + 14, o + 16384, 4096);  // g_w
    CV(b + 12, o + 20480, 4096);  // op_w
  }
  for (int pass = 0; pass < 2; ++pass) {
    int b = pass == 0 ? 34 : 44;
    size_t o = 49152 + pass * 20480;
    CV(b + 3, o + 0,     4096);   // wq
    CV(b + 4, o + 4096,  4096);   // wk
    CV(b + 5, o + 8192,  4096);   // wv
    CV(b + 6, o + 12288, 4096);   // wg
    CV(b + 8, o + 16384, 4096);   // wo
  }
  CV(56, 90112, 8192);            // w1
  CV(58, 98304, 8192);            // w2

  // ---- Triangle multiplication (outgoing: base=2, incoming: base=18) -----
  for (int pass = 0; pass < 2; ++pass) {
    int b = pass == 0 ? 2 : 18;
    int outgoing = pass == 0 ? 1 : 0;
    const __bf16* warena = wa + pass * 24576;
    ln_kernel<<<8192, 256, 0, stream>>>(act, F(b + 0), F(b + 1), a_bf, 0);
    tm_proj_kernel<<<512, 256, 0, stream>>>(a_bf, mask, warena,
        F(b + 3), F(b + 7),    // lp_b, lg_b
        F(b + 5), F(b + 9),    // rp_b, rg_b
        F(b + 15),             // g_b
        buf1, buf2, gatef, outgoing);
    // outgoing: A=left,B=right ; incoming: A=right,B=left (layouts arranged in tm_proj)
    tri_einsum_kernel<<<2048, 256, 0, stream>>>(outgoing ? buf1 : buf2,
                                                outgoing ? buf2 : buf1, ppf);
    ln_kernel<<<8192, 256, 0, stream>>>(ppf, F(b + 10), F(b + 11), a_bf, 0);
    gated_out_kernel<<<512, 256, 0, stream>>>(a_bf, warena + 20480, F(b + 13), gatef, act);
  }

  // ---- Triangle attention (start: base=34, t=0; end: base=44, t=1) -------
  for (int pass = 0; pass < 2; ++pass) {
    int b = pass == 0 ? 34 : 44;
    int t = pass;
    const __bf16* warena = wa + 49152 + pass * 20480;
    ln_kernel<<<8192, 256, 0, stream>>>(act, F(b + 0), F(b + 1), a_bf, t);
    nb_kernel<<<8192, 256, 0, stream>>>(a_bf, F(b + 2), nbf);
    attn_proj_kernel<<<512, 256, 0, stream>>>(a_bf, warena, F(b + 7),
                                              buf1, buf2, buf3, gatef);
    attn_kernel<<<NN * HH, 128, 0, stream>>>(buf1, buf2, buf3, gatef, nbf, mask, obuf, t);
    attn_out_kernel<<<512, 256, 0, stream>>>(obuf, warena + 16384, F(b + 9), act, t);
  }

  // ---- Transition (base=54) ----------------------------------------------
  ln_kernel<<<8192, 256, 0, stream>>>(act, F(54), F(55), a_bf, 0);
  trans1_kernel<<<512, 256, 0, stream>>>(a_bf, wa + 90112, F(57), hid);
  trans2_kernel<<<512, 256, 0, stream>>>(hid, wa + 98304, F(59), act);
}
